// NodeGCN_77197742178451
// MI455X (gfx1250) — compile-verified, matrix operand order checked by reading, not run on hardware
//
#include <hip/hip_runtime.h>

typedef __attribute__((ext_vector_type(16))) __bf16 v16bf;
typedef __attribute__((ext_vector_type(8)))  __bf16 v8bf;
typedef __attribute__((ext_vector_type(8)))  float  v8f;

#define IN_DIM 128
#define HIDDEN 128

// ---------------------------------------------------------------- degrees ---
__global__ void k_deg_init(float* __restrict__ deg, int n) {
    int i = blockIdx.x * blockDim.x + threadIdx.x;
    if (i < n) deg[i] = 1.0f;                 // self-loop contributes 1
}

__global__ void k_deg_count(const int* __restrict__ dst, float* __restrict__ deg, int e) {
    int i = blockIdx.x * blockDim.x + threadIdx.x;
    if (i < e) atomicAdd(&deg[dst[i]], 1.0f);
}

__global__ void k_dinv(float* __restrict__ deg, int n) {
    int i = blockIdx.x * blockDim.x + threadIdx.x;
    if (i < n) {
        float d = deg[i];
        deg[i] = (d > 0.0f) ? rsqrtf(d) : 0.0f;
    }
}

// ------------------------------------------------- WMMA GEMM: hw = x @ W1 ---
// One wave per 16-row tile, 8 waves per block. W1^T staged in LDS as bf16.
__global__ void __launch_bounds__(256) k_gemm1(const float* __restrict__ x,
                                               const float* __restrict__ W1,
                                               float* __restrict__ hw,
                                               int n_nodes) {
    __shared__ __bf16 lds_w[HIDDEN * IN_DIM];   // [n][k] = W1[k][n], 32 KB

    for (int idx = threadIdx.x; idx < IN_DIM * HIDDEN; idx += 256) {
        int k = idx >> 7, n = idx & 127;
        lds_w[n * IN_DIM + k] = (__bf16)W1[idx];
    }
    __syncthreads();

    const int wave = threadIdx.x >> 5;
    const int lane = threadIdx.x & 31;
    const int half = lane >> 4;                 // 0: lanes 0-15, 1: lanes 16-31
    const int m    = lane & 15;

    const int ntiles = (n_nodes + 15) >> 4;
    const int tile   = blockIdx.x * 8 + wave;
    if (tile >= ntiles) return;                 // wave-uniform: EXEC stays all-1s

    const int r0   = tile << 4;
    const bool full = (r0 + 16 <= n_nodes);     // uniform in practice (N%16==0)

    int row = r0 + m;
    if (row >= n_nodes) row = n_nodes - 1;      // only matters on ragged tail
    const float* xrow = x + (size_t)row * IN_DIM;

    // A: 4 K-tiles of 16x32 bf16 in the ISA 16-bit A layout.
    // lane<16 : VGPR0-3 = K[kt*32+0..7],  VGPR4-7 = K[kt*32+16..23]
    // lane>=16: VGPR0-3 = K[kt*32+8..15], VGPR4-7 = K[kt*32+24..31]
    v16bf a[4];
#pragma unroll
    for (int kt = 0; kt < 4; ++kt) {
        const int kb = kt * 32 + half * 8;
        float4 lo0 = *(const float4*)(xrow + kb);
        float4 lo1 = *(const float4*)(xrow + kb + 4);
        float4 hi0 = *(const float4*)(xrow + kb + 16);
        float4 hi1 = *(const float4*)(xrow + kb + 20);
        a[kt][0]  = (__bf16)lo0.x; a[kt][1]  = (__bf16)lo0.y;
        a[kt][2]  = (__bf16)lo0.z; a[kt][3]  = (__bf16)lo0.w;
        a[kt][4]  = (__bf16)lo1.x; a[kt][5]  = (__bf16)lo1.y;
        a[kt][6]  = (__bf16)lo1.z; a[kt][7]  = (__bf16)lo1.w;
        a[kt][8]  = (__bf16)hi0.x; a[kt][9]  = (__bf16)hi0.y;
        a[kt][10] = (__bf16)hi0.z; a[kt][11] = (__bf16)hi0.w;
        a[kt][12] = (__bf16)hi1.x; a[kt][13] = (__bf16)hi1.y;
        a[kt][14] = (__bf16)hi1.z; a[kt][15] = (__bf16)hi1.w;
    }

    for (int nt = 0; nt < 8; ++nt) {
        const int c = nt * 16 + m;              // output column this lane feeds
        const __bf16* wrow = &lds_w[c * IN_DIM];

        // Preload all four B K-tiles into independent registers so the 8
        // ds_load_b128 issue as a batch and the 4 WMMAs chain back-to-back.
        v16bf b[4];
#pragma unroll
        for (int kt = 0; kt < 4; ++kt) {
            const int kb = kt * 32 + half * 8;
            v8bf blo = *(const v8bf*)(wrow + kb);
            v8bf bhi = *(const v8bf*)(wrow + kb + 16);
#pragma unroll
            for (int i = 0; i < 8; ++i) { b[kt][i] = blo[i]; b[kt][8 + i] = bhi[i]; }
        }

        v8f acc = {};
#pragma unroll
        for (int kt = 0; kt < 4; ++kt)
            acc = __builtin_amdgcn_wmma_f32_16x16x32_bf16(
                false, a[kt], false, b[kt], (short)0, acc, false, false);

        // C/D f32 layout: VGPR v -> M = v + half*8, N = m
        float* dstp = hw + (size_t)(r0 + half * 8) * HIDDEN + nt * 16 + m;
        if (full) {                              // all-ones vcc on real tiles
#pragma unroll
            for (int v = 0; v < 8; ++v)
                dstp[(size_t)v * HIDDEN] = acc[v];
        } else {
#pragma unroll
            for (int v = 0; v < 8; ++v)
                if (r0 + half * 8 + v < n_nodes)
                    dstp[(size_t)v * HIDDEN] = acc[v];
        }
    }
}

// ------------------------------------- layer-1 aggregation (self-loop init) -
__global__ void k_agg1_init(const float* __restrict__ hw,
                            const float* __restrict__ dinv,
                            const float* __restrict__ b1,
                            float* __restrict__ agg, int n) {
    int t = blockIdx.x * blockDim.x + threadIdx.x;  // n*32 threads, 4 feats each
    int i = t >> 5;
    if (i >= n) return;
    int fq = (t & 31) * 4;
    float s = dinv[i]; s *= s;
    float4 v = *(const float4*)(hw + (size_t)i * HIDDEN + fq);
    float4 bb = *(const float4*)(b1 + fq);
    float4 o;
    o.x = v.x * s + bb.x; o.y = v.y * s + bb.y;
    o.z = v.z * s + bb.z; o.w = v.w * s + bb.w;
    *(float4*)(agg + (size_t)i * HIDDEN + fq) = o;
}

// ---------------------------------------- layer-1 per-edge feature scatter --
__global__ void k_edge1(const int* __restrict__ src, const int* __restrict__ dst,
                        const float* __restrict__ hw, const float* __restrict__ dinv,
                        float* __restrict__ agg, int e) {
    int t = blockIdx.x * blockDim.x + threadIdx.x;  // e*32 threads
    int ei = t >> 5;
    if (ei >= e) return;
    int fq = (t & 31) * 4;
    int s = src[ei], d = dst[ei];
    float nrm = dinv[s] * dinv[d];
    float4 v = *(const float4*)(hw + (size_t)s * HIDDEN + fq);
    float* o = agg + (size_t)d * HIDDEN + fq;
    atomicAdd(o + 0, v.x * nrm);
    atomicAdd(o + 1, v.y * nrm);
    atomicAdd(o + 2, v.z * nrm);
    atomicAdd(o + 3, v.w * nrm);
}

// ------------- fused ReLU + (h @ W2) dot + layer-2 self-loop/bias out init --
__global__ void __launch_bounds__(256) k_fuse2(const float* __restrict__ agg,
                                               const float* __restrict__ W2,
                                               const float* __restrict__ b2,
                                               const float* __restrict__ dinv,
                                               float* __restrict__ hw2,
                                               float* __restrict__ out, int n) {
    int wave = threadIdx.x >> 5, lane = threadIdx.x & 31;
    int i = blockIdx.x * 8 + wave;
    if (i >= n) return;
    int fq = lane * 4;
    float4 v = *(const float4*)(agg + (size_t)i * HIDDEN + fq);
    float4 w = *(const float4*)(W2 + fq);
    float p = fmaxf(v.x, 0.0f) * w.x + fmaxf(v.y, 0.0f) * w.y +
              fmaxf(v.z, 0.0f) * w.z + fmaxf(v.w, 0.0f) * w.w;
#pragma unroll
    for (int off = 16; off > 0; off >>= 1)
        p += __shfl_xor(p, off, 32);
    if (lane == 0) {
        float s = dinv[i];
        hw2[i] = p;
        out[i] = b2[0] + p * s * s;
    }
}

// ----------------------------------------- layer-2 per-edge scalar scatter --
__global__ void k_edge2(const int* __restrict__ src, const int* __restrict__ dst,
                        const float* __restrict__ hw2, const float* __restrict__ dinv,
                        float* __restrict__ out, int e) {
    int ei = blockIdx.x * blockDim.x + threadIdx.x;
    if (ei >= e) return;
    int s = src[ei], d = dst[ei];
    atomicAdd(&out[d], hw2[s] * dinv[s] * dinv[d]);
}

// ---------------------------------------------------------------- launcher --
extern "C" void kernel_launch(void* const* d_in, const int* in_sizes, int n_in,
                              void* d_out, int out_size, void* d_ws, size_t ws_size,
                              hipStream_t stream) {
    const float* x  = (const float*)d_in[0];
    const int*   ei = (const int*)d_in[1];
    const float* W1 = (const float*)d_in[2];
    const float* b1 = (const float*)d_in[3];
    const float* W2 = (const float*)d_in[4];
    const float* b2 = (const float*)d_in[5];
    float* out = (float*)d_out;

    const int N = in_sizes[0] / IN_DIM;   // 100000
    const int E = in_sizes[1] / 2;        // 1600000
    const int* src = ei;
    const int* dst = ei + E;

    char* ws = (char*)d_ws;
    float* hw   = (float*)ws; ws += (size_t)N * HIDDEN * sizeof(float);
    float* agg  = (float*)ws; ws += (size_t)N * HIDDEN * sizeof(float);
    float* dinv = (float*)ws; ws += (size_t)N * sizeof(float);
    float* hw2  = (float*)ws; ws += (size_t)N * sizeof(float);

    // degrees -> dinv (stored in `dinv`, reused as deg during accumulation)
    k_deg_init <<<(N + 255) / 256, 256, 0, stream>>>(dinv, N);
    k_deg_count<<<(E + 255) / 256, 256, 0, stream>>>(dst, dinv, E);
    k_dinv     <<<(N + 255) / 256, 256, 0, stream>>>(dinv, N);

    // hw = x @ W1 via bf16 WMMA
    const int ntiles = (N + 15) / 16;
    k_gemm1<<<(ntiles + 7) / 8, 256, 0, stream>>>(x, W1, hw, N);

    // layer-1 aggregate
    k_agg1_init<<<((size_t)N * 32 + 255) / 256, 256, 0, stream>>>(hw, dinv, b1, agg, N);
    k_edge1    <<<((size_t)E * 32 + 255) / 256, 256, 0, stream>>>(src, dst, hw, dinv, agg, E);

    // ReLU + W2 dot + layer-2 init, then layer-2 aggregate
    k_fuse2<<<(N + 7) / 8, 256, 0, stream>>>(agg, W2, b2, dinv, hw2, out, N);
    k_edge2<<<(E + 255) / 256, 256, 0, stream>>>(src, dst, hw2, dinv, out, E);
}
